// GINE_Net_13657996001716
// MI455X (gfx1250) — matrix operand order repaired and checked by reading, not compiled, not run on hardware
//
#include <hip/hip_runtime.h>

constexpr int DIM = 64;    // feature dim (in == hidden == out)
constexpr int ET  = 128;   // edges per block (4 waves x 32 rows)
constexpr int NT  = 128;   // nodes per block
constexpr int ROWP = 36;   // padded row stride (dwords) for packed bf16 tiles
constexpr int WFRAG = 2048; // dwords per packed weight matrix

typedef __attribute__((ext_vector_type(16))) __bf16 v16bf;
typedef __attribute__((ext_vector_type(8)))  float  v8f;

union BF8 { unsigned u[8]; uint4 q[2]; v16bf v; };

// round-to-nearest-even f32 -> bf16, packed pair (lo in [15:0], hi in [31:16])
__device__ __forceinline__ unsigned pack_bf16_2(float lo, float hi) {
  unsigned ul = __float_as_uint(lo);
  ul += 0x7FFFu + ((ul >> 16) & 1u);
  unsigned uh = __float_as_uint(hi);
  uh += 0x7FFFu + ((uh >> 16) & 1u);
  return (uh & 0xFFFF0000u) | (ul >> 16);
}

__device__ __forceinline__ uint2 pack_bf16_4(float4 t) {
  return make_uint2(pack_bf16_2(t.x, t.y), pack_bf16_2(t.z, t.w));
}

// A fragment from an LDS tile already packed bf16 (ROWP dwords/row).
__device__ __forceinline__ v16bf load_a_frag(const unsigned* sp, int row0, int kb,
                                             int lane) {
  const int r  = row0 + (lane & 15);
  const int p0 = (kb >> 1) + ((lane >> 4) & 1) * 4;
  const unsigned* p = sp + r * ROWP + p0;
  BF8 f;
  f.q[0] = *(const uint4*)p;
  f.q[1] = *(const uint4*)(p + 8);
  return f.v;
}

// B fragment from pre-packed global weight image (lane-stride 32B, coalesced).
__device__ __forceinline__ v16bf load_b_frag(const unsigned* wp, int kb, int n0,
                                             int lane) {
  const unsigned* p = wp + (((((kb >> 5) << 2) + n0) << 5) + lane) * 8;
  BF8 f;
  f.q[0] = *(const uint4*)p;
  f.q[1] = *(const uint4*)(p + 4);
  return f.v;
}

__device__ __forceinline__ v8f wmma_bf16(v16bf a, v16bf b, v8f c) {
  return __builtin_amdgcn_wmma_f32_16x16x32_bf16(false, a, false, b, (short)0, c,
                                                 false, false);
}

// ---------------- weight pre-pack: f32 [64x64] K-major -> fragment-ready bf16
__global__ void gine_pack_w_kernel(const float* __restrict__ W,
                                   unsigned* __restrict__ out) {
  const int idx = blockIdx.x * blockDim.x + threadIdx.x;
  if (idx >= WFRAG) return;
  const int v    = idx & 7;
  const int lane = (idx >> 3) & 31;
  const int n0   = (idx >> 8) & 3;
  const int kb   = (idx >> 10) & 1;
  const int k = kb * 32 + (v >> 2) * 16 + ((lane >> 4) & 1) * 8 + 2 * (v & 3);
  const int c = n0 * 16 + (lane & 15);
  out[idx] = pack_bf16_2(W[k * DIM + c], W[(k + 1) * DIM + c]);
}

// ---------------- edge kernel: e = ea@We + be; msg = relu(x[src]+e); agg[dst] += msg
__global__ __launch_bounds__(128) void gine_edge_kernel(
    const float* __restrict__ x, const int* __restrict__ src,
    const int* __restrict__ dst, const float* __restrict__ ea,
    const unsigned* __restrict__ Wp, const float* __restrict__ be,
    float* __restrict__ agg, int nE) {
  __shared__ __align__(16) unsigned sEp[ET * ROWP];
  __shared__ __align__(16) float sX[ET * DIM];
  __shared__ int sSrc[ET];
  __shared__ int sDst[ET];
  const int tid = threadIdx.x;
  const int e0  = blockIdx.x * ET;
  const bool full = (e0 + ET) <= nE;
  {
    const int e = e0 + tid;
    sSrc[tid] = (e < nE) ? src[e] : 0;
    sDst[tid] = (e < nE) ? dst[e] : -1;
  }
  __syncthreads();
  // stage edge tile as packed bf16; gather x[src] rows as f32 (float4 moves)
  for (int i = tid; i < ET * 16; i += 128) {
    const int r = i >> 4, q = i & 15;
    float4 ev = make_float4(0.f, 0.f, 0.f, 0.f);
    float4 xv = make_float4(0.f, 0.f, 0.f, 0.f);
    if (full || (e0 + r) < nE) {
      ev = *(const float4*)&ea[(size_t)(e0 + r) * DIM + 4 * q];
      xv = *(const float4*)&x[(size_t)sSrc[r] * DIM + 4 * q];
    }
    *(uint2*)&sEp[r * ROWP + 2 * q] = pack_bf16_4(ev);
    *(float4*)&sX[r * DIM + 4 * q] = xv;
  }
  __syncthreads();
  const int lane = tid & 31, row0 = (tid >> 5) * 32;
  v8f acc[2][4] = {};
#pragma unroll
  for (int kb = 0; kb < DIM; kb += 32) {
    const v16bf a0 = load_a_frag(sEp, row0, kb, lane);
    const v16bf a1 = load_a_frag(sEp, row0 + 16, kb, lane);
#pragma unroll
    for (int n = 0; n < 4; ++n) {
      const v16bf b = load_b_frag(Wp, kb, n, lane);
      acc[0][n] = wmma_bf16(a0, b, acc[0][n]);
      acc[1][n] = wmma_bf16(a1, b, acc[1][n]);
    }
  }
  const int col = lane & 15;
  const int hi8 = (lane >> 4) << 3;
  float bias[4];
#pragma unroll
  for (int n = 0; n < 4; ++n) bias[n] = be[n * 16 + col];
#pragma unroll
  for (int t = 0; t < 2; ++t) {
    if (full) {
#pragma unroll
      for (int r = 0; r < 8; ++r) {
        const int lr = row0 + t * 16 + hi8 + r;
        float* rowp = agg + (size_t)sDst[lr] * DIM + col;
        const float* xp = sX + lr * DIM + col;
#pragma unroll
        for (int n = 0; n < 4; ++n) {
          const float m = fmaxf(acc[t][n][r] + bias[n] + xp[n * 16], 0.f);
          atomicAdd(rowp + n * 16, m);
        }
      }
    } else {
#pragma unroll
      for (int r = 0; r < 8; ++r) {
        const int lr = row0 + t * 16 + hi8 + r;
        const int dn = sDst[lr];
        if (dn >= 0) {
          float* rowp = agg + (size_t)dn * DIM + col;
          const float* xp = sX + lr * DIM + col;
#pragma unroll
          for (int n = 0; n < 4; ++n) {
            const float m = fmaxf(acc[t][n][r] + bias[n] + xp[n * 16], 0.f);
            atomicAdd(rowp + n * 16, m);
          }
        }
      }
    }
  }
}

// ---------------- node1: h = (x+agg)@W1 + b1 (in-place over agg ok), BN sums
__global__ __launch_bounds__(128) void gine_node1_kernel(
    const float* __restrict__ x, const float* agg,
    const unsigned* __restrict__ Wp, const float* __restrict__ b1,
    float* h, float* __restrict__ stats, int nN) {
  __shared__ __align__(16) unsigned sAp[NT * ROWP];
  __shared__ float sS[2 * DIM];
  const int tid = threadIdx.x;
  const int n0  = blockIdx.x * NT;
  const bool full = (n0 + NT) <= nN;
  for (int i = tid; i < 2 * DIM; i += 128) sS[i] = 0.f;
  for (int i = tid; i < NT * 16; i += 128) {
    const int r = i >> 4, q = i & 15;
    float4 v = make_float4(0.f, 0.f, 0.f, 0.f);
    if (full || (n0 + r) < nN) {
      const float4 xv = *(const float4*)&x[(size_t)(n0 + r) * DIM + 4 * q];
      const float4 av = *(const float4*)&agg[(size_t)(n0 + r) * DIM + 4 * q];
      v = make_float4(xv.x + av.x, xv.y + av.y, xv.z + av.z, xv.w + av.w);
    }
    *(uint2*)&sAp[r * ROWP + 2 * q] = pack_bf16_4(v);
  }
  __syncthreads();
  const int lane = tid & 31, row0 = (tid >> 5) * 32;
  v8f acc[2][4] = {};
#pragma unroll
  for (int kb = 0; kb < DIM; kb += 32) {
    const v16bf a0 = load_a_frag(sAp, row0, kb, lane);
    const v16bf a1 = load_a_frag(sAp, row0 + 16, kb, lane);
#pragma unroll
    for (int n = 0; n < 4; ++n) {
      const v16bf b = load_b_frag(Wp, kb, n, lane);
      acc[0][n] = wmma_bf16(a0, b, acc[0][n]);
      acc[1][n] = wmma_bf16(a1, b, acc[1][n]);
    }
  }
  const int col = lane & 15;
  const int hi8 = (lane >> 4) << 3;
  float bias[4], ls[4] = {}, lq[4] = {};
#pragma unroll
  for (int n = 0; n < 4; ++n) bias[n] = b1[n * 16 + col];
#pragma unroll
  for (int t = 0; t < 2; ++t) {
#pragma unroll
    for (int r = 0; r < 8; ++r) {
      const int node = n0 + row0 + t * 16 + hi8 + r;
      if (full || node < nN) {
        float* rowp = h + (size_t)node * DIM + col;
#pragma unroll
        for (int n = 0; n < 4; ++n) {
          const float v = acc[t][n][r] + bias[n];
          rowp[n * 16] = v;
          ls[n] += v; lq[n] += v * v;
        }
      }
    }
  }
#pragma unroll
  for (int n = 0; n < 4; ++n) {
    atomicAdd(&sS[n * 16 + col], ls[n]);
    atomicAdd(&sS[DIM + n * 16 + col], lq[n]);
  }
  __syncthreads();
  for (int i = tid; i < 2 * DIM; i += 128) atomicAdd(&stats[i], sS[i]);
}

// ---------------- BN finalize: scale = g*rsqrt(var+eps); shift = bt - mu*scale
__global__ void gine_bn_kernel(const float* __restrict__ stats,
                               const float* __restrict__ g,
                               const float* __restrict__ bt,
                               float* __restrict__ ss, float invN) {
  const int c = threadIdx.x;
  const float mu  = stats[c] * invN;
  const float var = stats[DIM + c] * invN - mu * mu;
  const float sc  = g[c] * rsqrtf(var + 1e-5f);
  ss[c]       = sc;
  ss[DIM + c] = bt[c] - mu * sc;
}

// ---------------- node2: out = relu(h*scale+shift) @ W2 + b2 (+ optional relu)
__global__ __launch_bounds__(128) void gine_node2_kernel(
    const float* __restrict__ h, const float* __restrict__ ss,
    const unsigned* __restrict__ Wp, const float* __restrict__ b2,
    float* __restrict__ out, int nN, int finalRelu) {
  __shared__ __align__(16) unsigned sAp[NT * ROWP];
  __shared__ float sSS[2 * DIM];
  const int tid = threadIdx.x;
  const int n0  = blockIdx.x * NT;
  const bool full = (n0 + NT) <= nN;
  if (tid < 2 * DIM) sSS[tid] = ss[tid];
  __syncthreads();
  for (int i = tid; i < NT * 16; i += 128) {
    const int r = i >> 4, q = i & 15;
    float4 v = make_float4(0.f, 0.f, 0.f, 0.f);
    if (full || (n0 + r) < nN) {
      const float4 hv = *(const float4*)&h[(size_t)(n0 + r) * DIM + 4 * q];
      const int c0 = 4 * q;
      v.x = fmaxf(hv.x * sSS[c0 + 0] + sSS[DIM + c0 + 0], 0.f);
      v.y = fmaxf(hv.y * sSS[c0 + 1] + sSS[DIM + c0 + 1], 0.f);
      v.z = fmaxf(hv.z * sSS[c0 + 2] + sSS[DIM + c0 + 2], 0.f);
      v.w = fmaxf(hv.w * sSS[c0 + 3] + sSS[DIM + c0 + 3], 0.f);
    }
    *(uint2*)&sAp[r * ROWP + 2 * q] = pack_bf16_4(v);
  }
  __syncthreads();
  const int lane = tid & 31, row0 = (tid >> 5) * 32;
  v8f acc[2][4] = {};
#pragma unroll
  for (int kb = 0; kb < DIM; kb += 32) {
    const v16bf a0 = load_a_frag(sAp, row0, kb, lane);
    const v16bf a1 = load_a_frag(sAp, row0 + 16, kb, lane);
#pragma unroll
    for (int n = 0; n < 4; ++n) {
      const v16bf b = load_b_frag(Wp, kb, n, lane);
      acc[0][n] = wmma_bf16(a0, b, acc[0][n]);
      acc[1][n] = wmma_bf16(a1, b, acc[1][n]);
    }
  }
  const int col = lane & 15;
  const int hi8 = (lane >> 4) << 3;
  float bias[4];
#pragma unroll
  for (int n = 0; n < 4; ++n) bias[n] = b2[n * 16 + col];
#pragma unroll
  for (int t = 0; t < 2; ++t) {
#pragma unroll
    for (int r = 0; r < 8; ++r) {
      const int node = n0 + row0 + t * 16 + hi8 + r;
      if (full || node < nN) {
        float* rowp = out + (size_t)node * DIM + col;
#pragma unroll
        for (int n = 0; n < 4; ++n) {
          float o = acc[t][n][r] + bias[n];
          if (finalRelu) o = fmaxf(o, 0.f);
          rowp[n * 16] = o;
        }
      }
    }
  }
}

__global__ void gine_zero_kernel(float4* __restrict__ p, int n4) {
  const int i = blockIdx.x * blockDim.x + threadIdx.x;
  if (i < n4) p[i] = make_float4(0.f, 0.f, 0.f, 0.f);
}

extern "C" void kernel_launch(void* const* d_in, const int* in_sizes, int n_in,
                              void* d_out, int out_size, void* d_ws, size_t ws_size,
                              hipStream_t stream) {
  const float* x   = (const float*)d_in[0];
  const int*  eidx = (const int*)d_in[1];
  const float* ea  = (const float*)d_in[2];
  const int N = in_sizes[0] / DIM;
  const int E = in_sizes[1] / 2;
  const int* srcp = eidx;
  const int* dstp = eidx + E;

  const float* We[2] = {(const float*)d_in[3],  (const float*)d_in[11]};
  const float* be[2] = {(const float*)d_in[4],  (const float*)d_in[12]};
  const float* W1[2] = {(const float*)d_in[5],  (const float*)d_in[13]};
  const float* b1[2] = {(const float*)d_in[6],  (const float*)d_in[14]};
  const float* g[2]  = {(const float*)d_in[7],  (const float*)d_in[15]};
  const float* bt[2] = {(const float*)d_in[8],  (const float*)d_in[16]};
  const float* W2[2] = {(const float*)d_in[9],  (const float*)d_in[17]};
  const float* b2[2] = {(const float*)d_in[10], (const float*)d_in[18]};

  float* buf0  = (float*)d_ws;                 // agg, then h (in place)
  float* buf1  = buf0 + (size_t)N * DIM;       // inter-layer activations
  float* stats = buf1 + (size_t)N * DIM;       // 2*DIM sums
  float* ss    = stats + 2 * DIM;              // 2*DIM scale/shift
  unsigned* wpk = (unsigned*)(ss + 2 * DIM);   // 6 * WFRAG dwords packed weights

  const int nND = N * DIM;
  const dim3 zgrid((nND / 4 + 255) / 256);
  const dim3 egrid((E + ET - 1) / ET);
  const dim3 ngrid((N + NT - 1) / NT);
  const dim3 wgrid((WFRAG + 255) / 256);
  const float invN = 1.0f / (float)N;

  // pre-pack all 6 weight matrices into fragment-ready bf16 images
  for (int l = 0; l < 2; ++l) {
    gine_pack_w_kernel<<<wgrid, 256, 0, stream>>>(We[l], wpk + (3 * l + 0) * WFRAG);
    gine_pack_w_kernel<<<wgrid, 256, 0, stream>>>(W1[l], wpk + (3 * l + 1) * WFRAG);
    gine_pack_w_kernel<<<wgrid, 256, 0, stream>>>(W2[l], wpk + (3 * l + 2) * WFRAG);
  }

  const float* xin = x;
  for (int l = 0; l < 2; ++l) {
    gine_zero_kernel<<<zgrid, 256, 0, stream>>>((float4*)buf0, nND / 4);
    gine_zero_kernel<<<1, 64, 0, stream>>>((float4*)stats, (2 * DIM) / 4);
    gine_edge_kernel<<<egrid, 128, 0, stream>>>(xin, srcp, dstp, ea,
                                                wpk + (3 * l + 0) * WFRAG, be[l],
                                                buf0, E);
    gine_node1_kernel<<<ngrid, 128, 0, stream>>>(xin, buf0,
                                                 wpk + (3 * l + 1) * WFRAG, b1[l],
                                                 buf0, stats, N);
    gine_bn_kernel<<<1, DIM, 0, stream>>>(stats, g[l], bt[l], ss, invN);
    float* outp = (l == 0) ? buf1 : (float*)d_out;
    gine_node2_kernel<<<ngrid, 128, 0, stream>>>(buf0, ss,
                                                 wpk + (3 * l + 2) * WFRAG, b2[l],
                                                 outp, N, (l == 0) ? 1 : 0);
    xin = buf1;
  }
}
